// MinLSTMCell_901943132455
// MI455X (gfx1250) — matrix operand order repaired and checked by reading, not compiled, not run on hardware
//
#include <hip/hip_runtime.h>
#include <hip/hip_bf16.h>
#include <math.h>

typedef __bf16 bf16_t;
typedef bf16_t v16bf __attribute__((ext_vector_type(16)));
typedef bf16_t v8bf  __attribute__((ext_vector_type(8)));
typedef float  v8f   __attribute__((ext_vector_type(8)));
typedef unsigned int u32x4 __attribute__((ext_vector_type(4)));
typedef int i32x4 __attribute__((ext_vector_type(4)));
typedef int i32x8 __attribute__((ext_vector_type(8)));

#define B_ 256
#define T_ 512
#define D_ 256
#define U_ 256
#define G_ 768   // 3*U

__device__ __forceinline__ float sigmoidf_(float x) {
    return 1.0f / (1.0f + __expf(-x));
}

// ---------------------------------------------------------------------------
// TDM: 1D async copy of 24 KB (12288 bf16) global -> LDS.
// D# per CDNA5 ISA ch.8: group0 {count=1, lds_addr, global_addr, type=2},
// group1 {data_size=2B, tensor_dim0=12288, tensor_dim1=1, tile_dim0=12288,
//         tile_dim1=1, dim0_stride=12288}.
// ---------------------------------------------------------------------------
__device__ __forceinline__ void tdm_load_xw(const bf16_t* gsrc, unsigned lds_off) {
    unsigned long long ga = (unsigned long long)(uintptr_t)gsrc;
    u32x4 g0 = { 1u,                                   // count=1, user mode
                 lds_off,                              // lds_addr (bits 63:32)
                 (unsigned)ga,                         // global_addr lo
                 (unsigned)((ga >> 32) & 0x1FFFFFFull) | 0x80000000u }; // type=2
    i32x8 g1 = { 0x00010000,                           // data_size=1 (2 bytes)
                 0x30000000,                           // tensor_dim0[15:0]=12288 << 16
                 0x00010000,                           // dim0 hi=0 | tensor_dim1=1
                 0x30000000,                           // tile_dim0=12288 << 16
                 1,                                    // tile_dim1=1, tile_dim2=0
                 12288,                                // tensor_dim0_stride lo
                 0, 0 };
    i32x4 gz = { 0, 0, 0, 0 };
#if __clang_major__ >= 23
    i32x8 gz8 = { 0, 0, 0, 0, 0, 0, 0, 0 };
    __builtin_amdgcn_tensor_load_to_lds(g0, g1, gz, gz, gz8, 0);
#else
    __builtin_amdgcn_tensor_load_to_lds(g0, g1, gz, gz, 0);
#endif
}

// ---------------------------------------------------------------------------
// Pack a logical [256 x 768] matrix (three 256x256 row-major blocks side by
// side) into WMMA bf16 B-fragments.
// Fragment (kt, nt): 32 lanes x 16 elems contiguous.
//   elem e of lane L = M[32*kt + (e&7) + 8*(L/16) + 16*(e/8)][16*nt + (L%16)]
// ---------------------------------------------------------------------------
__global__ void pack_b_kernel(const float* __restrict__ S0,
                              const float* __restrict__ S1,
                              const float* __restrict__ S2,
                              bf16_t* __restrict__ dst) {
    int tid = blockIdx.x * blockDim.x + threadIdx.x;
    if (tid >= 8 * 48 * 512) return;
    int e  = tid & 15;
    int L  = (tid >> 4) & 31;
    int f  = tid >> 9;
    int kt = f / 48;
    int nt = f - kt * 48;
    int K  = kt * 32 + (e & 7) + ((L >= 16) ? 8 : 0) + ((e >= 8) ? 16 : 0);
    int n  = nt * 16 + (L & 15);
    const float* S = (n < 256) ? S0 : ((n < 512) ? S1 : S2);
    dst[tid] = (bf16_t)S[K * 256 + (n & 255)];
}

// ---------------------------------------------------------------------------
// xw = x @ [Wf|Wi|Wc] + b, M iterated over r2 = t*256 + b, output stored as
// C-fragment-layout bf16: xwpk[((t*16 + bblk)*48 + ntile)*256 + lane*8 + r]
// grid (M/128, 4), block 256 (8 waves). Wave w: 16 M-rows, 12 N tiles.
// 96 KB LDS for W fragments; 4-way N split halves redundant x reads vs 8-way.
// ---------------------------------------------------------------------------
__global__ void __launch_bounds__(256)
gemm_xw_kernel(const float* __restrict__ x,
               const bf16_t* __restrict__ Wpk,
               const float* __restrict__ bf_,
               const float* __restrict__ bi_,
               const float* __restrict__ bc_,
               bf16_t* __restrict__ xwpk) {
    __shared__ bf16_t sW[8 * 12 * 512];   // 96 KB: 96 B-fragments for this N slice

    const int tid   = threadIdx.x;
    const int lane  = tid & 31;
    const int w     = tid >> 5;
    const int Nbase = blockIdx.y * 192;
    const int m0    = blockIdx.x * 128 + w * 16;  // r2 base of this wave's tile

    // cooperatively stage W fragments (kt 0..7, 12 local N tiles) into LDS
    {
        uint4* d4 = (uint4*)sW;
        const uint4* s4 = (const uint4*)Wpk;
        #pragma unroll 4
        for (int i = tid; i < 8 * 12 * 64; i += 256) {
            int fl  = i >> 6;            // local fragment 0..95
            int off = i & 63;
            int kt  = fl / 12;
            int ntl = fl - kt * 12;
            d4[i] = s4[((size_t)kt * 48 + (Nbase >> 4) + ntl) * 64 + off];
        }
    }
    __syncthreads();

    // accumulators initialized with bias
    v8f acc[12];
    #pragma unroll
    for (int j = 0; j < 12; ++j) {
        int n = Nbase + j * 16 + (lane & 15);
        const float* bsrc = (n < 256) ? bf_ : ((n < 512) ? bi_ : bc_);
        float bv = bsrc[n & 255];
        #pragma unroll
        for (int r = 0; r < 8; ++r) acc[j][r] = bv;
    }

    // A row for this lane: r2 = m0 + lane%16 -> t = r2>>8, b = r2&255
    const int r2   = m0 + (lane & 15);
    const int hi8  = (lane >= 16) ? 8 : 0;
    const float* arow = x + ((size_t)(r2 & 255) * T_ + (r2 >> 8)) * D_ + hi8;

    #pragma unroll
    for (int kt = 0; kt < 8; ++kt) {
        // build bf16 A-fragment on the fly (two contiguous 8-float chunks)
        const float4* ap = (const float4*)(arow + kt * 32);
        float4 c0v = ap[0], c1v = ap[1], c2v = ap[4], c3v = ap[5];
        float tmp[16] = {c0v.x, c0v.y, c0v.z, c0v.w, c1v.x, c1v.y, c1v.z, c1v.w,
                         c2v.x, c2v.y, c2v.z, c2v.w, c3v.x, c3v.y, c3v.z, c3v.w};
        v16bf a;
        #pragma unroll
        for (int e = 0; e < 16; ++e) a[e] = (bf16_t)tmp[e];

        #pragma unroll
        for (int j = 0; j < 12; ++j) {
            v16bf bfr = *(const v16bf*)(sW + (kt * 12 + j) * 512 + lane * 16);
            acc[j] = __builtin_amdgcn_wmma_f32_16x16x32_bf16(
                false, a, false, bfr, (short)0, acc[j], false, false);
        }
    }

    // packed store: whole tile shares t = m0>>8 and batch block (m0&255)>>4
    const size_t fragbase = ((size_t)(m0 >> 8) * 16 + ((m0 & 255) >> 4)) * 48;
    #pragma unroll
    for (int j = 0; j < 12; ++j) {
        size_t frag = fragbase + (Nbase >> 4) + j;
        v8bf o;
        #pragma unroll
        for (int r = 0; r < 8; ++r) o[r] = (bf16_t)acc[j][r];
        *(v8bf*)(xwpk + frag * 256 + lane * 8) = o;   // 16B store per lane
    }
}

// ---------------------------------------------------------------------------
// Recurrent scan. grid 16 (B/16), block 512 (16 waves).
// Wave w owns units [16w,16w+16): gate tiles {w, 16+w, 32+w}. Uh fragments
// live in VGPRs for the whole scan (24 x v16bf per wave). h fragments are
// double-buffered in LDS and written directly in packed layout (no staging,
// ONE barrier per step). xw double-buffered into LDS by the TDM.
// ---------------------------------------------------------------------------
__global__ void __launch_bounds__(512)
scan_kernel(const bf16_t* __restrict__ xwpk,
            const bf16_t* __restrict__ Upk,
            const float* __restrict__ h0,
            const float* __restrict__ c0,
            float* __restrict__ out) {
    __shared__ bf16_t sXW[2 * 12288];   // 48 KB xw double buffer (TDM dest)
    __shared__ bf16_t sA[2 * 4096];     // 16 KB double-buffered packed h frags

    const int tid  = threadIdx.x;
    const int lane = tid & 31;
    const int w    = tid >> 5;          // 0..15
    const int b0   = blockIdx.x * 16;
    const int Mhi  = (lane >= 16) ? 8 : 0;
    const int u    = w * 16 + (lane & 15);

    // scatter coordinates for writing h(m,u) into A-fragment layout:
    // frag kt=u>>5; k'=u&31; elem e=(k'&7)|(k'>=16?8:0); lane-half=(k'>>3)&1
    const int ktU   = u >> 5;
    const int kpr   = u & 31;
    const int eU    = (kpr & 7) | ((kpr >= 16) ? 8 : 0);
    const int halfU = (kpr >> 3) & 1;
    const int aOff  = ktU * 512 + (Mhi + 16 * halfU) * 16 + eU;  // + r*16

    // ---- initial h -> packed A fragments in buffer 0 ----
    if (tid < 256) {
        int kt = tid >> 5, L = tid & 31;
        int m = L & 15, h8 = (L >= 16) ? 8 : 0;
        const float* hr = h0 + (size_t)(b0 + m) * U_ + kt * 32 + h8;
        v16bf a;
        #pragma unroll
        for (int e = 0; e < 8; ++e)  a[e] = (bf16_t)hr[e];
        #pragma unroll
        for (int e = 8; e < 16; ++e) a[e] = (bf16_t)hr[e + 8];
        *(v16bf*)(sA + kt * 512 + L * 16) = a;
    }

    // ---- register-resident Uh fragments (3 gate tiles x 8 k-steps) ----
    v16bf uhf[8], uhi[8], uhc[8];
    #pragma unroll
    for (int kt = 0; kt < 8; ++kt) {
        uhf[kt] = *(const v16bf*)(Upk + ((size_t)kt * 48 + w)      * 512 + lane * 16);
        uhi[kt] = *(const v16bf*)(Upk + ((size_t)kt * 48 + 16 + w) * 512 + lane * 16);
        uhc[kt] = *(const v16bf*)(Upk + ((size_t)kt * 48 + 32 + w) * 512 + lane * 16);
    }

    // ---- cell state (16 units x 16 rows, accumulator layout) ----
    v8f c1;
    #pragma unroll
    for (int r = 0; r < 8; ++r)
        c1[r] = c0[(size_t)(b0 + r + Mhi) * U_ + u];

    const int bblk = blockIdx.x;
    const unsigned ldsXW0 = (unsigned)(uintptr_t)(&sXW[0]);

    // prologue: TDM-load xw for t=0 into buffer 0
    if (w == 0) {
        tdm_load_xw(xwpk + (size_t)(0 * 16 + bblk) * 48 * 256, ldsXW0);
        __builtin_amdgcn_s_wait_tensorcnt(0);
    }
    __syncthreads();

    for (int t = 0; t < T_; ++t) {
        // kick off async TDM for t+1 into the other buffer
        if (w == 0 && t + 1 < T_)
            tdm_load_xw(xwpk + (size_t)((t + 1) * 16 + bblk) * 48 * 256,
                        ldsXW0 + (unsigned)(((t + 1) & 1) * 24576));

        // accumulators start from xw tile in LDS (bias already folded in)
        const bf16_t* bufp = sXW + (t & 1) * 12288;
        v8f af, ai, ac;
        {
            v8bf xf = *(const v8bf*)(bufp + (w)      * 256 + lane * 8);
            v8bf xi = *(const v8bf*)(bufp + (16 + w) * 256 + lane * 8);
            v8bf xc = *(const v8bf*)(bufp + (32 + w) * 256 + lane * 8);
            #pragma unroll
            for (int r = 0; r < 8; ++r) {
                af[r] = (float)xf[r];
                ai[r] = (float)xi[r];
                ac[r] = (float)xc[r];
            }
        }

        // gates += h @ Uh (weights in registers, h fragments from LDS)
        const bf16_t* aP = sA + (t & 1) * 4096;
        #pragma unroll
        for (int kt = 0; kt < 8; ++kt) {
            v16bf a = *(const v16bf*)(aP + kt * 512 + lane * 16);
            af = __builtin_amdgcn_wmma_f32_16x16x32_bf16(
                false, a, false, uhf[kt], (short)0, af, false, false);
            ai = __builtin_amdgcn_wmma_f32_16x16x32_bf16(
                false, a, false, uhi[kt], (short)0, ai, false, false);
            ac = __builtin_amdgcn_wmma_f32_16x16x32_bf16(
                false, a, false, uhc[kt], (short)0, ac, false, false);
        }

        // elementwise LSTM update; write h to output and directly into the
        // next step's packed A fragments (other parity buffer)
        bf16_t* aN = sA + (((t + 1) & 1) * 4096) + aOff;
        #pragma unroll
        for (int r = 0; r < 8; ++r) {
            float f  = sigmoidf_(af[r]);
            float ii = sigmoidf_(ai[r]);
            float cc = tanhf(ac[r]);
            float cn = f * c1[r] + ii * cc;
            c1[r]    = cn;
            float hn = tanhf(cn);
            int m = r + Mhi;
            out[((size_t)(b0 + m) * T_ + t) * U_ + u] = hn;
            aN[r * 16] = (bf16_t)hn;
        }

        if (w == 0 && t + 1 < T_)
            __builtin_amdgcn_s_wait_tensorcnt(0);   // xw for t+1 landed
        __syncthreads();   // h frags + xw buffer ready for next step
    }
}

// ---------------------------------------------------------------------------
extern "C" void kernel_launch(void* const* d_in, const int* in_sizes, int n_in,
                              void* d_out, int out_size, void* d_ws, size_t ws_size,
                              hipStream_t stream) {
    (void)in_sizes; (void)n_in; (void)out_size; (void)ws_size;
    const float* x  = (const float*)d_in[0];
    const float* Wf = (const float*)d_in[1];
    const float* Uf = (const float*)d_in[2];
    const float* bf = (const float*)d_in[3];
    const float* Wi = (const float*)d_in[4];
    const float* Ui = (const float*)d_in[5];
    const float* bi = (const float*)d_in[6];
    const float* Wc = (const float*)d_in[7];
    const float* Uc = (const float*)d_in[8];
    const float* bc = (const float*)d_in[9];
    const float* h0 = (const float*)d_in[10];
    const float* c0 = (const float*)d_in[11];
    float* out = (float*)d_out;

    char* ws = (char*)d_ws;
    bf16_t* Wpk  = (bf16_t*)ws;                    // 384 KB packed W
    bf16_t* Upk  = (bf16_t*)(ws + 384 * 1024);     // 384 KB packed Uh
    bf16_t* xwpk = (bf16_t*)(ws + (1 << 20));      // 192 MB packed xw fragments

    pack_b_kernel<<<768, 256, 0, stream>>>(Wf, Wi, Wc, Wpk);
    pack_b_kernel<<<768, 256, 0, stream>>>(Uf, Ui, Uc, Upk);
    gemm_xw_kernel<<<dim3(1024, 4), 256, 0, stream>>>(x, Wpk, bf, bi, bc, xwpk);
    scan_kernel<<<16, 512, 0, stream>>>(xwpk, Upk, h0, c0, out);
}
